// UltraMem_94489280805
// MI455X (gfx1250) — matrix operand order repaired and checked by reading, not compiled, not run on hardware
//
#include <hip/hip_runtime.h>
#include <hip/hip_bf16.h>

typedef __attribute__((ext_vector_type(16))) __bf16 v16bf;
typedef __attribute__((ext_vector_type(8)))  float  v8f;

#define BATCH 2
#define SEQ   2048
#define BN    4096          // BATCH*SEQ
#define DIM   1024
#define DQK   128
#define DVAL  256
#define NKEYS 1000
#define GKEYS 4000          // 4 * NKEYS (h0row,h1row,h0col,h1col)
#define TOPK  32
#define NEG_INF (-3.402823466e38f)

// ---------------------------------------------------------------- K0: 2x2 SVD
// uv[0..3] = u_vec (h0x,h0y,h1x,h1y); uv[4..7] = t_vec
__global__ void svd2x2_kernel(const float* __restrict__ core, float* __restrict__ uv) {
  if (threadIdx.x != 0 || blockIdx.x != 0) return;
  for (int h = 0; h < 2; ++h) {
    float a = core[h*4+0], b = core[h*4+1], c = core[h*4+2], d = core[h*4+3];
    // A^T A = [[al, be],[be, ga]]
    float al = a*a + c*c, be = a*b + c*d, ga = b*b + d*d;
    float df   = 0.5f*(al - ga);
    float lam  = 0.5f*(al + ga) + sqrtf(df*df + be*be);
    float e1x = be,        e1y = lam - al;
    float e2x = lam - ga,  e2y = be;
    float vx, vy;
    if (e1x*e1x + e1y*e1y > e2x*e2x + e2y*e2y) { vx = e1x; vy = e1y; }
    else                                        { vx = e2x; vy = e2y; }
    float nv = sqrtf(vx*vx + vy*vy);
    if (nv < 1e-30f) { vx = 1.f; vy = 0.f; } else { vx /= nv; vy /= nv; }
    float ux = a*vx + b*vy, uy = c*vx + d*vy;
    float nu = sqrtf(ux*ux + uy*uy);
    if (nu < 1e-30f) { ux = 1.f; uy = 0.f; } else { ux /= nu; uy /= nu; }
    uv[h*2+0] = ux; uv[h*2+1] = uy;      // u_vec[h]
    uv[4+h*2+0] = vx; uv[4+h*2+1] = vy;  // t_vec[h]
  }
}

// ---------------------------------------------------------------- K1: rmsnorm
__global__ __launch_bounds__(256) void rmsnorm_kernel(const float* __restrict__ tokens,
                                                      const float* __restrict__ g,
                                                      float* __restrict__ out) {
  __shared__ float red[256];
  int row = blockIdx.x;                       // b*SEQ + n
  const float* x = tokens + (size_t)row * DIM;
  float*       o = out    + (size_t)row * DIM;
  float ss = 0.f;
  for (int d = threadIdx.x; d < DIM; d += 256) { float v = x[d]; ss += v * v; }
  red[threadIdx.x] = ss;
  __syncthreads();
  for (int s = 128; s > 0; s >>= 1) {
    if (threadIdx.x < s) red[threadIdx.x] += red[threadIdx.x + s];
    __syncthreads();
  }
  float inv = rsqrtf(red[0] / (float)DIM + 1e-6f);
  for (int d = threadIdx.x; d < DIM; d += 256) o[d] = x[d] * inv * g[d];
}

// ------------------------------------------------- K2: causal conv3 -> bf16
__global__ __launch_bounds__(256) void conv_kernel(const float* __restrict__ xn,
                                                   const float* __restrict__ w,
                                                   const float* __restrict__ bias,
                                                   __bf16* __restrict__ out) {
  int idx = blockIdx.x * 256 + threadIdx.x;   // < BN*DIM
  int d = idx & (DIM - 1);
  int n = (idx >> 10) & (SEQ - 1);
  int b = idx >> 21;
  const float* base = xn + (size_t)b * SEQ * DIM;
  float v2 = base[(size_t)n * DIM + d];
  float v1 = (n >= 1) ? base[(size_t)(n - 1) * DIM + d] : 0.f;
  float v0 = (n >= 2) ? base[(size_t)(n - 2) * DIM + d] : 0.f;
  float r = v0 * w[d*3+0] + v1 * w[d*3+1] + v2 * w[d*3+2] + bias[d];
  out[idx] = (__bf16)r;
}

// ---------------------------------------------------------- K2b: f32 -> bf16
__global__ __launch_bounds__(256) void cvt_bf16_kernel(const float* __restrict__ in,
                                                       __bf16* __restrict__ out, int n) {
  int idx = blockIdx.x * 256 + threadIdx.x;
  if (idx < n) out[idx] = (__bf16)in[idx];
}

// -------------------------------------------- K2c: rank-merged keys -> bf16
// mk layout [g][m][k], g: 0=h0row 1=h1row 2=h0col 3=h1col
__global__ __launch_bounds__(256) void mkkeys_kernel(const float* __restrict__ keys,
                                                     const float* __restrict__ uv,
                                                     __bf16* __restrict__ mk) {
  int idx = blockIdx.x * 256 + threadIdx.x;   // < 4*NKEYS*DQK
  int k = idx & (DQK - 1);
  int m = (idx >> 7) % NKEYS;
  int g = idx / (NKEYS * DQK);
  int h = g & 1, cc = g >> 1;                 // cc: 0=row(keys[0]) 1=col(keys[1])
  float w0, w1;
  if (cc == 0) { w0 = uv[h*2+0];   w1 = uv[h*2+1];   }
  else         { w0 = uv[4+h*2+0]; w1 = uv[4+h*2+1]; }
  size_t base = (size_t)cc * 2 * NKEYS * DQK;
  float v = w0 * keys[base + (size_t)m * DQK + k]
          + w1 * keys[base + (size_t)NKEYS * DQK + (size_t)m * DQK + k];
  mk[idx] = (__bf16)v;
}

// ------------------------------------------------ WMMA bf16 GEMM: C = A * B^T
// A: MxK row-major bf16 (lda=K); B: NxK row-major bf16 (ldb=K); C: MxN f32.
// Each wave computes a 16x32 C strip (1 A frag reused across 2 WMMAs / k-step).
// K multiple of 32; M multiple of 16; N multiple of 32.
__global__ __launch_bounds__(256) void gemm_bf16_wmma(const __bf16* __restrict__ A,
                                                      const __bf16* __restrict__ Bm,
                                                      float* __restrict__ C,
                                                      int M, int N, int K, int numJobs) {
  int wave = threadIdx.x >> 5;
  int lane = threadIdx.x & 31;
  int job  = blockIdx.x * 8 + wave;
  if (job >= numJobs) return;                 // wave-uniform
  int stripsN = N >> 5;                       // 32-wide strips
  int tm = (job / stripsN) << 4;
  int tn = (job % stripsN) << 5;
  int  l15 = lane & 15;
  bool hi  = lane >= 16;
  // A 16x32 bf16 frag: lanes0-15 K0..15, lanes16-31 K16..31
  const __bf16* Arow  = A  + (size_t)(tm + l15) * K + (hi ? 16 : 0);
  // B 32x16 bf16 frag: lanes0-15 K0..15, lanes16-31 K16..31 (col = lane&15)
  const __bf16* Brow0 = Bm + (size_t)(tn + l15) * K + (hi ? 16 : 0);
  const __bf16* Brow1 = Brow0 + (size_t)16 * K;
  v8f c0 = {0.f, 0.f, 0.f, 0.f, 0.f, 0.f, 0.f, 0.f};
  v8f c1 = {0.f, 0.f, 0.f, 0.f, 0.f, 0.f, 0.f, 0.f};
  for (int k0 = 0; k0 < K; k0 += 32) {
    if (k0 + 64 < K) {                        // speculative prefetch 2 steps ahead
      __builtin_prefetch(Arow  + k0 + 64, 0, 1);
      __builtin_prefetch(Brow0 + k0 + 64, 0, 1);
      __builtin_prefetch(Brow1 + k0 + 64, 0, 1);
    }
    v16bf a, b0, b1;
#pragma unroll
    for (int j = 0; j < 8; ++j)  a[j]     = Arow[k0 + j];
#pragma unroll
    for (int j = 0; j < 8; ++j)  a[8 + j] = Arow[k0 + 8 + j];
#pragma unroll
    for (int j = 0; j < 16; ++j) b0[j]    = Brow0[k0 + j];
#pragma unroll
    for (int j = 0; j < 16; ++j) b1[j]    = Brow1[k0 + j];
    c0 = __builtin_amdgcn_wmma_f32_16x16x32_bf16(false, a, false, b0, (short)0, c0,
                                                 false, false);
    c1 = __builtin_amdgcn_wmma_f32_16x16x32_bf16(false, a, false, b1, (short)0, c1,
                                                 false, false);
  }
  // C 16x16 f32: VGPR j -> M=j (lanes0-15) / M=j+8 (lanes16-31), N=lane&15
  int cn  = tn + l15;
  int cm0 = tm + (hi ? 8 : 0);
#pragma unroll
  for (int j = 0; j < 8; ++j) C[(size_t)(cm0 + j) * N + cn]      = c0[j];
#pragma unroll
  for (int j = 0; j < 8; ++j) C[(size_t)(cm0 + j) * N + cn + 16] = c1[j];
}

// ------------------------------------------- K4: layernorm q (+bf16 copy)
__global__ __launch_bounds__(128) void lnq_kernel(const float* __restrict__ qpre,
                                                  const float* __restrict__ gw,
                                                  float* __restrict__ qf,
                                                  __bf16* __restrict__ qb) {
  __shared__ float s1[4], s2[4];
  int row = blockIdx.x;
  float v = qpre[(size_t)row * DQK + threadIdx.x];
  float sv = v, sq = v * v;
#pragma unroll
  for (int o = 16; o > 0; o >>= 1) { sv += __shfl_xor(sv, o, 32); sq += __shfl_xor(sq, o, 32); }
  int w = threadIdx.x >> 5, l = threadIdx.x & 31;
  if (l == 0) { s1[w] = sv; s2[w] = sq; }
  __syncthreads();
  float m   = (s1[0] + s1[1] + s1[2] + s1[3]) / (float)DQK;
  float var = (s2[0] + s2[1] + s2[2] + s2[3]) / (float)DQK - m * m;
  float r = (v - m) * rsqrtf(var + 1e-5f) * gw[threadIdx.x];
  qf[(size_t)row * DQK + threadIdx.x] = r;
  qb[(size_t)row * DQK + threadIdx.x] = (__bf16)r;
}

// ----------------------------------------- K6: top-32 of 1000 (per h,rc,tok)
// blockIdx.x = g*BN + token, g = rc*2+h; merged layout [token][g*NKEYS+m]
__global__ __launch_bounds__(256) void topk1000_kernel(const float* __restrict__ merged,
                                                       int* __restrict__ topidx) {
  __shared__ float vals[NKEYS];
  __shared__ float rv[256];
  __shared__ int   ri[256];
  int g     = blockIdx.x >> 12;
  int token = blockIdx.x & (BN - 1);
  int t = threadIdx.x;
  const float* src = merged + (size_t)token * GKEYS + (size_t)g * NKEYS;
  for (int m = t; m < NKEYS; m += 256) vals[m] = src[m];
  __syncthreads();
  for (int it = 0; it < TOPK; ++it) {
    float bv = NEG_INF; int bi = 0x7fffffff;
    for (int m = t; m < NKEYS; m += 256) {
      float v = vals[m];
      if (v > bv) { bv = v; bi = m; }
    }
    rv[t] = bv; ri[t] = bi;
    __syncthreads();
    for (int s = 128; s > 0; s >>= 1) {
      if (t < s) {
        float ov = rv[t + s]; int oi = ri[t + s];
        if (ov > rv[t] || (ov == rv[t] && oi < ri[t])) { rv[t] = ov; ri[t] = oi; }
      }
      __syncthreads();
    }
    if (t == 0) {
      int wi = ri[0];
      topidx[(size_t)blockIdx.x * TOPK + it] = wi;
      vals[wi] = NEG_INF;
    }
    __syncthreads();
  }
}

// ----- K7: cross scores, top-32 of 1024, sigmoid, gather mem_table, reduce
__global__ __launch_bounds__(256) void final_kernel(const float* __restrict__ qf,
                                                    const float* __restrict__ keys,
                                                    const float* __restrict__ core,
                                                    const int* __restrict__ topidx,
                                                    const float* __restrict__ mem,
                                                    float* __restrict__ out) {
  __shared__ float qs[DQK];
  __shared__ float frc[128];        // fr[32][2] at 0..63, fc[32][2] at 64..127
  __shared__ float sval[1024];
  __shared__ float rv[256];
  __shared__ int   ri[256];
  __shared__ int   rowi[TOPK], coli[TOPK];
  __shared__ float wsel;
  __shared__ int   msel;
  int token = blockIdx.x;
  int t = threadIdx.x;
  if (t < DQK) qs[t] = qf[(size_t)token * DQK + t];
  float acc = 0.f;
  int wv = t >> 5, ln = t & 31;
  for (int h = 0; h < 2; ++h) {
    if (t < TOPK)           rowi[t]       = topidx[((size_t)h       * BN + token) * TOPK + t];
    else if (t < 2 * TOPK)  coli[t - 32]  = topidx[((size_t)(2 + h) * BN + token) * TOPK + (t - 32)];
    __syncthreads();
    // 128 dot products of length 128: fr (64) then fc (64)
    for (int dd = wv; dd < 128; dd += 8) {
      int which = dd >> 6;            // 0 = fr (keys c=0), 1 = fc (keys c=1)
      int i = (dd & 63) >> 1;
      int r = dd & 1;
      int mi = (which == 0) ? rowi[i] : coli[i];
      const float* kp = keys + (((size_t)which * 2 + r) * NKEYS + mi) * DQK;
      float s = 0.f;
      for (int kk = ln; kk < DQK; kk += 32) s += qs[kk] * kp[kk];
#pragma unroll
      for (int o = 16; o > 0; o >>= 1) s += __shfl_xor(s, o, 32);
      if (ln == 0) frc[dd] = s;
    }
    __syncthreads();
    float c00 = core[h*4+0], c01 = core[h*4+1], c10 = core[h*4+2], c11 = core[h*4+3];
    for (int e = t; e < 1024; e += 256) {
      int i = e >> 5, j = e & 31;
      float f0 = frc[i*2+0],      f1 = frc[i*2+1];
      float g0 = frc[64 + j*2+0], g1 = frc[64 + j*2+1];
      sval[e] = f0 * (c00 * g0 + c01 * g1) + f1 * (c10 * g0 + c11 * g1);
    }
    __syncthreads();
    for (int it = 0; it < TOPK; ++it) {
      float bv = NEG_INF; int bi = 0x7fffffff;
      for (int e = t; e < 1024; e += 256) {
        float v = sval[e];
        if (v > bv) { bv = v; bi = e; }
      }
      rv[t] = bv; ri[t] = bi;
      __syncthreads();
      for (int s = 128; s > 0; s >>= 1) {
        if (t < s) {
          float ov = rv[t + s]; int oi = ri[t + s];
          if (ov > rv[t] || (ov == rv[t] && oi < ri[t])) { rv[t] = ov; ri[t] = oi; }
        }
        __syncthreads();
      }
      if (t == 0) {
        int e = ri[0];
        sval[e] = NEG_INF;
        msel = rowi[e >> 5] * NKEYS + coli[e & 31];
        wsel = 1.f / (1.f + __expf(-rv[0]));
      }
      __syncthreads();
      acc += wsel * mem[(size_t)msel * DVAL + t];
      __syncthreads();
    }
    __syncthreads();
  }
  out[(size_t)token * DVAL + t] = acc;
}

// --------------------------------------------------------------------- host
static inline size_t align256(size_t x) { return (x + 255) & ~(size_t)255; }

extern "C" void kernel_launch(void* const* d_in, const int* in_sizes, int n_in,
                              void* d_out, int out_size, void* d_ws, size_t ws_size,
                              hipStream_t stream) {
  const float* tokens  = (const float*)d_in[0];
  const float* g_rms   = (const float*)d_in[1];
  const float* conv_w  = (const float*)d_in[2];
  const float* conv_b  = (const float*)d_in[3];
  const float* Wq      = (const float*)d_in[4];
  const float* q_ln_w  = (const float*)d_in[5];
  /* d_in[6] = k_ln_w : unused by the reference forward */
  const float* keys    = (const float*)d_in[7];
  const float* core    = (const float*)d_in[8];
  const float* memtab  = (const float*)d_in[9];
  float* out = (float*)d_out;

  // workspace carve (merged overlays xnorm: xnorm dead before merged written)
  char* ws = (char*)d_ws;
  size_t off = 0;
  float*  merged = (float*)(ws + off);                 // BN*GKEYS f32
  float*  xnorm  = (float*)(ws + off);                 // BN*DIM f32 (overlay)
  off += align256((size_t)BN * GKEYS * 4);
  __bf16* xconv  = (__bf16*)(ws + off); off += align256((size_t)BN * DIM * 2);
  __bf16* wq_bf  = (__bf16*)(ws + off); off += align256((size_t)DQK * DIM * 2);
  __bf16* mk     = (__bf16*)(ws + off); off += align256((size_t)GKEYS * DQK * 2);
  float*  qpre   = (float*)(ws + off);  off += align256((size_t)BN * DQK * 4);
  float*  qf     = (float*)(ws + off);  off += align256((size_t)BN * DQK * 4);
  __bf16* qbf    = (__bf16*)(ws + off); off += align256((size_t)BN * DQK * 2);
  int*    topidx = (int*)(ws + off);    off += align256((size_t)4 * BN * TOPK * 4);
  float*  uv     = (float*)(ws + off);  off += 256;
  (void)ws_size; (void)in_sizes; (void)n_in; (void)out_size;

  svd2x2_kernel<<<1, 32, 0, stream>>>(core, uv);
  rmsnorm_kernel<<<BN, 256, 0, stream>>>(tokens, g_rms, xnorm);
  conv_kernel<<<(BN * DIM) / 256, 256, 0, stream>>>(xnorm, conv_w, conv_b, xconv);
  cvt_bf16_kernel<<<(DQK * DIM) / 256, 256, 0, stream>>>(Wq, wq_bf, DQK * DIM);
  mkkeys_kernel<<<(4 * NKEYS * DQK) / 256, 256, 0, stream>>>(keys, uv, mk);
  // GEMM1: qpre[BN,DQK] = xconv[BN,DIM] * Wq^T   (16x32 strips)
  {
    int numJobs = (BN / 16) * (DQK / 32);              // 1024
    gemm_bf16_wmma<<<numJobs / 8, 256, 0, stream>>>(xconv, wq_bf, qpre,
                                                    BN, DQK, DIM, numJobs);
  }
  lnq_kernel<<<BN, DQK, 0, stream>>>(qpre, q_ln_w, qf, qbf);
  // GEMM2: merged[BN,GKEYS] = q[BN,DQK] * mk^T   (16x32 strips)
  {
    int numJobs = (BN / 16) * (GKEYS / 32);            // 32000
    gemm_bf16_wmma<<<numJobs / 8, 256, 0, stream>>>(qbf, mk, merged,
                                                    BN, GKEYS, DQK, numJobs);
  }
  topk1000_kernel<<<4 * BN, 256, 0, stream>>>(merged, topidx);
  final_kernel<<<BN, 256, 0, stream>>>(qf, keys, core, topidx, memtab, out);
}
